// CrossCovarianceAttention_15659450761282
// MI455X (gfx1250) — compile-verified, matrix operand order checked by reading, not run on hardware
//
#include <hip/hip_runtime.h>
#include <hip/hip_bf16.h>
#include <math.h>

// ---------------------------------------------------------------------------
// CrossCovarianceAttention (XCA) for MI455X / gfx1250, wave32 + WMMA bf16.
// All four matmuls use v_wmma_f32_16x16x32_bf16 (fp32 accumulate).
// Norm scales + temperature are folded into the attention logits.
// ---------------------------------------------------------------------------

typedef __bf16 bf16_t;
typedef __attribute__((ext_vector_type(16))) __bf16 v16bf;
typedef __attribute__((ext_vector_type(8)))  __bf16 v8bf;
typedef __attribute__((ext_vector_type(8)))  float  v8f;

#define HEADS   12
#define DH      64
#define DMODEL  768
#define D3      2304
#define NTOK    4096
#define BATCH   8
#define MTOT    (BATCH * NTOK)   // 32768 tokens total
#define EPSN    1e-12f

// ---------------- WMMA helpers ----------------------------------------------

__device__ inline v8f wmma_bf16(v16bf a, v16bf b, v8f c) {
  // D = A(16x32) x B(32x16) + C, fp32 accum
  return __builtin_amdgcn_wmma_f32_16x16x32_bf16(
      /*neg_a=*/false, a, /*neg_b=*/false, b,
      /*c_mod=*/(short)0, c, /*reuse_a=*/false, /*reuse_b=*/false);
}

// A-fragment (16x32, bf16, row-major, stride ld in elements).
// Per ISA: lanes 0-15 hold row M=lane, K = 0..7 then 16..23;
//          lanes 16-31 hold row M=lane-16, K = 8..15 then 24..31.
__device__ inline v16bf frag_bf16(const bf16_t* __restrict__ tile, int ld) {
  const int lane = threadIdx.x & 31;
  const int r    = lane & 15;
  const int k0   = (lane >> 4) << 3;           // 0 or 8
  const bf16_t* p = tile + (size_t)r * ld + k0;
  v8bf lo = *(const v8bf*)p;                   // K = k0 .. k0+7
  v8bf hi = *(const v8bf*)(p + 16);            // K = k0+16 .. k0+23
  v16bf f;
#pragma unroll
  for (int i = 0; i < 8; ++i) { f[i] = lo[i]; f[i + 8] = hi[i]; }
  return f;
}

__device__ inline v8f vzero8() {
  v8f z = {0.f, 0.f, 0.f, 0.f, 0.f, 0.f, 0.f, 0.f};
  return z;
}

__device__ inline v8bf pack8(const v8f& a) {
  v8bf o;
#pragma unroll
  for (int i = 0; i < 8; ++i) o[i] = (bf16_t)a[i];
  return o;
}

// ---------------- Kernel 0a: elementwise fp32 -> bf16 (for x) ---------------
__global__ void convert_to_bf16(const float* __restrict__ in,
                                bf16_t* __restrict__ out, long n8) {
  long i = (long)blockIdx.x * blockDim.x + threadIdx.x;  // 8 elems per thread
  if (i >= n8) return;
  const float4* p = (const float4*)(in + i * 8);
  float4 a = p[0], b = p[1];
  v8bf o;
  o[0] = (bf16_t)a.x; o[1] = (bf16_t)a.y; o[2] = (bf16_t)a.z; o[3] = (bf16_t)a.w;
  o[4] = (bf16_t)b.x; o[5] = (bf16_t)b.y; o[6] = (bf16_t)b.z; o[7] = (bf16_t)b.w;
  *(v8bf*)(out + i * 8) = o;
}

// ---------------- Kernel 0b: transpose + fp32 -> bf16 (weights) -------------
// out[c*R + r] = in[r*C + c]
__global__ void transpose_to_bf16(const float* __restrict__ in,
                                  bf16_t* __restrict__ out, int R, int C) {
  long i = (long)blockIdx.x * blockDim.x + threadIdx.x;
  long total = (long)R * C;
  if (i >= total) return;
  int r = (int)(i / C);
  int c = (int)(i % C);
  out[(long)c * R + r] = (bf16_t)in[i];
}

// ---------------- Kernel 1: qkv = x @ Wqkv ----------------------------------
// M = 32768 tokens, N = 2304 features, K = 768.
// Block = 4 waves (2x2); each wave owns a 32x64 tile (2x4 WMMA accumulators,
// 8 WMMA per k-step). A wave's 64-wide feature strip never crosses a q/k/v
// section boundary, so the destination branch is wave-uniform. q,k stored
// channel-major (B*768 channels x 4096 tokens) with vectorized v8bf stores;
// v kept in natural token-major layout.
__global__ __launch_bounds__(128) void qkv_gemm(
    const bf16_t* __restrict__ xb, const bf16_t* __restrict__ WqkvT,
    bf16_t* __restrict__ qT, bf16_t* __restrict__ kT,
    bf16_t* __restrict__ vbuf) {
  const int wave = threadIdx.x >> 5;
  const int lane = threadIdx.x & 31;
  const int m0 = blockIdx.y * 64 + (wave >> 1) * 32;    // token tile origin
  const int f0 = blockIdx.x * 128 + (wave & 1) * 64;    // feature tile origin

  v8f acc[2][4];
#pragma unroll
  for (int mi = 0; mi < 2; ++mi)
#pragma unroll
    for (int ni = 0; ni < 4; ++ni) acc[mi][ni] = vzero8();

  for (int kk = 0; kk < DMODEL; kk += 32) {
    v16bf a0 = frag_bf16(xb + (size_t)m0 * DMODEL + kk, DMODEL);
    v16bf a1 = frag_bf16(xb + (size_t)(m0 + 16) * DMODEL + kk, DMODEL);
#pragma unroll
    for (int ni = 0; ni < 4; ++ni) {
      v16bf b = frag_bf16(WqkvT + (size_t)(f0 + 16 * ni) * DMODEL + kk, DMODEL);
      acc[0][ni] = wmma_bf16(a0, b, acc[0][ni]);
      acc[1][ni] = wmma_bf16(a1, b, acc[1][ni]);
    }
  }

  const int nLoc = lane & 15;
  const int mOff = (lane >> 4) * 8;
  const int sec  = f0 / DMODEL;                  // wave-uniform: 0=q,1=k,2=v
  if (sec < 2) {
    bf16_t* dst = (sec == 0) ? qT : kT;
    const int fbase = f0 - sec * DMODEL;
#pragma unroll
    for (int mi = 0; mi < 2; ++mi) {
      int tokBase = m0 + 16 * mi + mOff;         // 8 consecutive tokens
      int b = tokBase >> 12;
      int n = tokBase & 4095;
#pragma unroll
      for (int ni = 0; ni < 4; ++ni) {
        int f = fbase + 16 * ni + nLoc;          // channel
        *(v8bf*)(dst + ((size_t)b * DMODEL + f) * NTOK + n) = pack8(acc[mi][ni]);
      }
    }
  } else {
    const int fbase = f0 - 2 * DMODEL;
#pragma unroll
    for (int mi = 0; mi < 2; ++mi)
#pragma unroll
      for (int ni = 0; ni < 4; ++ni) {
        int f = fbase + 16 * ni + nLoc;
#pragma unroll
        for (int i = 0; i < 8; ++i) {
          int tok = m0 + 16 * mi + mOff + i;
          vbuf[(size_t)tok * DMODEL + f] = (bf16_t)acc[mi][ni][i];
        }
      }
  }
}

// ---------------- Kernel 2: per-channel L2 norms over tokens ----------------
// One block per channel (b*768 + h*64 + c). sq = 1/max(||q||,eps);
// sk = temperature[h]/max(||k||,eps)  (temperature folded in here).
__global__ __launch_bounds__(256) void norms_kernel(
    const bf16_t* __restrict__ qT, const bf16_t* __restrict__ kT,
    const float* __restrict__ temperature,
    float* __restrict__ sq, float* __restrict__ sk) {
  const int ch = blockIdx.x;                 // 0 .. B*768-1
  const int h  = (ch % DMODEL) / DH;
  __shared__ float rq[256], rk[256];
  const bf16_t* qp = qT + (size_t)ch * NTOK;
  const bf16_t* kp = kT + (size_t)ch * NTOK;
  float aq = 0.f, ak = 0.f;
  for (int n = threadIdx.x; n < NTOK; n += 256) {
    float q = (float)qp[n]; aq += q * q;
    float k = (float)kp[n]; ak += k * k;
  }
  rq[threadIdx.x] = aq; rk[threadIdx.x] = ak;
  __syncthreads();
  for (int s = 128; s > 0; s >>= 1) {
    if ((int)threadIdx.x < s) {
      rq[threadIdx.x] += rq[threadIdx.x + s];
      rk[threadIdx.x] += rk[threadIdx.x + s];
    }
    __syncthreads();
  }
  if (threadIdx.x == 0) {
    sq[ch] = 1.f / fmaxf(sqrtf(rq[0]), EPSN);
    sk[ch] = temperature[h] / fmaxf(sqrtf(rk[0]), EPSN);
  }
}

// ---------------- Kernel 3: G = K^T Q (per b,h), scale, softmax -------------
// One block (4 waves) per (b,h). Wave w computes rows p = 16w..16w+15 of the
// 64x64 Gram matrix with inner dim N=4096 (contiguous thanks to channel-major
// storage). Softmax over q-axis in LDS; result stored transposed (q,p) bf16.
__global__ __launch_bounds__(128) void attn_kernel(
    const bf16_t* __restrict__ qT, const bf16_t* __restrict__ kT,
    const float* __restrict__ sq, const float* __restrict__ sk,
    bf16_t* __restrict__ attnT) {
  const int bh   = blockIdx.x;               // 0..95
  const int wave = threadIdx.x >> 5;
  const int lane = threadIdx.x & 31;
  __shared__ float sA[64 * 64];

  const bf16_t* Kb = kT + (size_t)bh * DH * NTOK;
  const bf16_t* Qb = qT + (size_t)bh * DH * NTOK;
  const int p0 = wave * 16;

  v8f acc[4];
#pragma unroll
  for (int j = 0; j < 4; ++j) acc[j] = vzero8();

  for (int n = 0; n < NTOK; n += 32) {
    v16bf a = frag_bf16(Kb + (size_t)p0 * NTOK + n, NTOK);
#pragma unroll
    for (int j = 0; j < 4; ++j) {
      v16bf b = frag_bf16(Qb + (size_t)(j * 16) * NTOK + n, NTOK);
      acc[j] = wmma_bf16(a, b, acc[j]);
    }
  }

  const int qLoc = lane & 15;
  const int pOff = (lane >> 4) * 8;
  const float* skp = sk + bh * DH;
  const float* sqp = sq + bh * DH;
#pragma unroll
  for (int j = 0; j < 4; ++j)
#pragma unroll
    for (int i = 0; i < 8; ++i) {
      int p = p0 + pOff + i;
      int q = j * 16 + qLoc;
      sA[p * 64 + q] = acc[j][i] * skp[p] * sqp[q];
    }
  __syncthreads();

  if (threadIdx.x < 64) {
    const int p = threadIdx.x;
    float m = -1e30f;
    for (int q = 0; q < 64; ++q) m = fmaxf(m, sA[p * 64 + q]);
    float s = 0.f;
    for (int q = 0; q < 64; ++q) s += __expf(sA[p * 64 + q] - m);
    float inv = 1.f / s;
    for (int q = 0; q < 64; ++q)
      attnT[(size_t)bh * 4096 + q * 64 + p] =
          (bf16_t)(__expf(sA[p * 64 + q] - m) * inv);
  }
}

// ---------------- Kernel 4: out_h = V @ attn, recombine heads ---------------
// Per (b,h): M = 4096 tokens x N = 64 x K = 64. Wave owns 32x64 tile.
__global__ __launch_bounds__(128) void av_kernel(
    const bf16_t* __restrict__ vbuf, const bf16_t* __restrict__ attnT,
    bf16_t* __restrict__ tmp) {
  const int bh   = blockIdx.x;
  const int b    = bh / HEADS;
  const int h    = bh % HEADS;
  const int wave = threadIdx.x >> 5;
  const int lane = threadIdx.x & 31;
  const int t0   = blockIdx.y * 128 + wave * 32;  // token within batch

  const bf16_t* Ab = vbuf + ((size_t)b * NTOK + t0) * DMODEL + h * DH;
  const bf16_t* Bb = attnT + (size_t)bh * 4096;

  v8f acc[2][4];
#pragma unroll
  for (int mi = 0; mi < 2; ++mi)
#pragma unroll
    for (int j = 0; j < 4; ++j) acc[mi][j] = vzero8();

#pragma unroll
  for (int kk = 0; kk < DH; kk += 32) {
    v16bf a0 = frag_bf16(Ab + kk, DMODEL);
    v16bf a1 = frag_bf16(Ab + (size_t)16 * DMODEL + kk, DMODEL);
#pragma unroll
    for (int j = 0; j < 4; ++j) {
      v16bf bfr = frag_bf16(Bb + (size_t)(j * 16) * 64 + kk, 64);
      acc[0][j] = wmma_bf16(a0, bfr, acc[0][j]);
      acc[1][j] = wmma_bf16(a1, bfr, acc[1][j]);
    }
  }

  const int qLoc = lane & 15;
  const int mOff = (lane >> 4) * 8;
#pragma unroll
  for (int mi = 0; mi < 2; ++mi)
#pragma unroll
    for (int j = 0; j < 4; ++j)
#pragma unroll
      for (int i = 0; i < 8; ++i) {
        int t = t0 + 16 * mi + mOff + i;
        int q = j * 16 + qLoc;
        tmp[((size_t)b * NTOK + t) * DMODEL + h * DH + q] =
            (bf16_t)acc[mi][j][i];
      }
}

// ---------------- Kernel 5: out = tmp @ Wout + bout -------------------------
// Wave owns a 32x64 tile (2x4 accumulators).
__global__ __launch_bounds__(128) void out_gemm(
    const bf16_t* __restrict__ tmp, const bf16_t* __restrict__ WoutT,
    const float* __restrict__ bout, float* __restrict__ out) {
  const int wave = threadIdx.x >> 5;
  const int lane = threadIdx.x & 31;
  const int m0 = blockIdx.y * 64 + (wave >> 1) * 32;
  const int f0 = blockIdx.x * 128 + (wave & 1) * 64;

  v8f acc[2][4];
#pragma unroll
  for (int mi = 0; mi < 2; ++mi)
#pragma unroll
    for (int ni = 0; ni < 4; ++ni) acc[mi][ni] = vzero8();

  for (int kk = 0; kk < DMODEL; kk += 32) {
    v16bf a0 = frag_bf16(tmp + (size_t)m0 * DMODEL + kk, DMODEL);
    v16bf a1 = frag_bf16(tmp + (size_t)(m0 + 16) * DMODEL + kk, DMODEL);
#pragma unroll
    for (int ni = 0; ni < 4; ++ni) {
      v16bf b = frag_bf16(WoutT + (size_t)(f0 + 16 * ni) * DMODEL + kk, DMODEL);
      acc[0][ni] = wmma_bf16(a0, b, acc[0][ni]);
      acc[1][ni] = wmma_bf16(a1, b, acc[1][ni]);
    }
  }

  const int nLoc = lane & 15;
  const int mOff = (lane >> 4) * 8;
#pragma unroll
  for (int mi = 0; mi < 2; ++mi)
#pragma unroll
    for (int ni = 0; ni < 4; ++ni) {
      int f = f0 + 16 * ni + nLoc;
      float bias = bout[f];
#pragma unroll
      for (int i = 0; i < 8; ++i) {
        int tok = m0 + 16 * mi + mOff + i;
        out[(size_t)tok * DMODEL + f] = acc[mi][ni][i] + bias;
      }
    }
}

// ---------------- Host-side launcher ----------------------------------------

extern "C" void kernel_launch(void* const* d_in, const int* in_sizes, int n_in,
                              void* d_out, int out_size, void* d_ws,
                              size_t ws_size, hipStream_t stream) {
  (void)in_sizes; (void)n_in; (void)out_size; (void)ws_size;
  const float* x           = (const float*)d_in[0];
  const float* Wqkv        = (const float*)d_in[1];
  const float* temperature = (const float*)d_in[2];
  const float* Wout        = (const float*)d_in[3];
  const float* bout        = (const float*)d_in[4];
  float*       out         = (float*)d_out;

  char*  ws  = (char*)d_ws;
  size_t off = 0;
  auto alloc = [&](size_t bytes) -> char* {
    char* p = ws + off;
    off = (off + bytes + 255) & ~(size_t)255;
    return p;
  };

  bf16_t* WqkvT = (bf16_t*)alloc((size_t)D3 * DMODEL * sizeof(bf16_t));
  bf16_t* WoutT = (bf16_t*)alloc((size_t)DMODEL * DMODEL * sizeof(bf16_t));
  bf16_t* qT    = (bf16_t*)alloc((size_t)BATCH * DMODEL * NTOK * sizeof(bf16_t));
  bf16_t* kT    = (bf16_t*)alloc((size_t)BATCH * DMODEL * NTOK * sizeof(bf16_t));
  bf16_t* vbuf  = (bf16_t*)alloc((size_t)MTOT * DMODEL * sizeof(bf16_t));
  float*  sq    = (float*)alloc((size_t)BATCH * DMODEL * sizeof(float));
  float*  sk    = (float*)alloc((size_t)BATCH * DMODEL * sizeof(float));
  bf16_t* attnT = (bf16_t*)alloc((size_t)BATCH * HEADS * DH * DH * sizeof(bf16_t));
  // xb (x in bf16) and tmp have disjoint live ranges -> share one buffer.
  bf16_t* xb    = (bf16_t*)alloc((size_t)MTOT * DMODEL * sizeof(bf16_t));
  bf16_t* tmpb  = xb;

  {
    long n8 = (long)MTOT * DMODEL / 8;
    convert_to_bf16<<<(unsigned)((n8 + 255) / 256), 256, 0, stream>>>(x, xb, n8);
  }
  {
    long tot = (long)DMODEL * D3;
    transpose_to_bf16<<<(unsigned)((tot + 255) / 256), 256, 0, stream>>>(
        Wqkv, WqkvT, DMODEL, D3);
  }
  {
    long tot = (long)DMODEL * DMODEL;
    transpose_to_bf16<<<(unsigned)((tot + 255) / 256), 256, 0, stream>>>(
        Wout, WoutT, DMODEL, DMODEL);
  }

  qkv_gemm<<<dim3(D3 / 128, MTOT / 64), 128, 0, stream>>>(xb, WqkvT, qT, kT, vbuf);

  norms_kernel<<<BATCH * DMODEL, 256, 0, stream>>>(qT, kT, temperature, sq, sk);

  attn_kernel<<<BATCH * HEADS, 128, 0, stream>>>(qT, kT, sq, sk, attnT);

  av_kernel<<<dim3(BATCH * HEADS, NTOK / 128), 128, 0, stream>>>(vbuf, attnT, tmpb);

  out_gemm<<<dim3(DMODEL / 128, MTOT / 64), 128, 0, stream>>>(tmpb, WoutT, bout, out);
}
